// RefineNet_74002286510442
// MI455X (gfx1250) — matrix operand order repaired and checked by reading, not compiled, not run on hardware
//
#include <hip/hip_runtime.h>

// CDNA5 (gfx1250) fused RefineNet implementation.
// conv1 (3x3, CIN->32) runs on the WMMA pipe (v_wmma_f32_16x16x32_f16),
// hidden tile kept in LDS, conv2 (3x3, 32->1) consumed from LDS.
// Each wave owns 3 M-tiles -> 6 independent WMMA accumulator chains, which
// covers the f16 WMMA->WMMA data hazard (5 slots) with useful work.

typedef __attribute__((ext_vector_type(16))) _Float16 v16h;
typedef __attribute__((ext_vector_type(8)))  _Float16 v8h;
typedef __attribute__((ext_vector_type(8)))  float    v8f;

namespace {
constexpr int Bb = 8, Nn = 14, Hh = 256, Ww = 256;
constexpr int Tt = 7;                 // REFINE_ITERS
constexpr int NP = 13;                // N-1 pairs
constexpr int PLANE = Hh * Ww;        // 65536
constexpr int FP_ELEMS = Bb * Nn * PLANE;          // 7,340,032
constexpr int RES_BSTRIDE = Tt * NP * PLANE;       // per-b stride in residual
}

// Flat offset (in halves) of im2col element k inside the padded feature tile:
// sFeat is [CIN_PAD][24][20]; k = c*9 + ty*3 + tx.  Rows 20..23 are zero pad
// so over-shot hidden pixels (p >= 324) read 0.0 (never NaN).
constexpr int foff(int k) {
  return (k / 9) * 480 + ((k % 9) / 3) * 20 + ((k % 9) % 3);
}

__device__ __forceinline__ float feat_pair(int c, int gy, int gx,
    const float* Ic, const float* In_, const float* e0, const float* e1,
    const float* pr) {
  if (gy < 0 || gy >= Hh || gx < 0 || gx >= Ww) return 0.f;   // 'SAME' zero pad
  int o = gy * Ww + gx;
  switch (c) {
    case 0: return Ic[o];
    case 1: return In_[o];
    case 2: return (gx == 0) ? 0.f : (Ic[o - 1]  - Ic[o]);    // I_x
    case 3: return (gy == 0) ? 0.f : (Ic[o - Ww] - Ic[o]);    // I_y
    case 4: return (gx == 0) ? 0.f : (In_[o - 1]  - In_[o]);  // Ixn
    case 5: return (gy == 0) ? 0.f : (In_[o - Ww] - In_[o]);  // Iyn
    case 6: return e0[o];
    case 7: return e1[o];
    default: return pr[o];                                    // prev residual
  }
}

__device__ __forceinline__ float relu_(float v) { return v > 0.f ? v : 0.f; }

// Fused conv1(WMMA)+ReLU+conv2 over a 16x16 output tile.
// CIN = 8 (first pair), 9 (other pairs), 1 (final head).
template <int CIN>
__global__ __launch_bounds__(256) void fused_conv(
    const float* __restrict__ src, const float* __restrict__ event,
    const float* __restrict__ prevres, float* __restrict__ outp,
    const float* __restrict__ w1, const float* __restrict__ w2,
    int plane_i, int bstride)
{
  constexpr int KTOT    = CIN * 9;
  constexpr int NCHUNK  = (KTOT + 31) / 32;
  constexpr int KPAD    = NCHUNK * 32;
  constexpr int CIN_PAD = (KPAD + 8) / 9;   // covers c = (KPAD-1)/9

  // All LDS zero-padded so WMMA-feeding gathers need NO bounds predication.
  __shared__ __align__(16) _Float16 sFeat[CIN_PAD * 24 * 20]; // halo-2 features
  __shared__ __align__(16) _Float16 sW1T[32][KPAD];           // conv1 w, [n][k]
  __shared__ __align__(16) _Float16 sHid[384 * 32];           // hidden, [pix][c]
  __shared__ float                  sW2[9][32];               // conv2 w [tap][c]

  const int tid = threadIdx.x;
  const int oh = blockIdx.y * 16, ow = blockIdx.x * 16;

  const float* Ic;
  const float* In_ = nullptr; const float* e0 = nullptr;
  const float* e1 = nullptr;  const float* pr = nullptr;
  float* out;
  if (CIN == 1) {                                    // final head: z = image id
    Ic  = src  + (size_t)blockIdx.z * PLANE;
    out = outp + (size_t)blockIdx.z * PLANE;
  } else {                                           // pair conv: z = batch
    int b = blockIdx.z;
    Ic  = src + ((size_t)b * Nn + plane_i) * PLANE;
    In_ = Ic + PLANE;
    e0  = event + ((size_t)b * (2 * (Nn - 1)) + 2 * plane_i) * PLANE;
    e1  = e0 + PLANE;
    if (CIN == 9) pr = prevres + (size_t)b * bstride;
    out = outp + (size_t)b * bstride;
  }

  // ---- stage conv1 weights n-major [n][k], zero-padded in k ----
  for (int idx = tid; idx < 32 * KPAD; idx += 256) {
    int n = idx / KPAD, k = idx % KPAD;
    float v = (k < KTOT) ? w1[n * KTOT + k] : 0.f;   // OIHW is contiguous in k
    sW1T[n][k] = (_Float16)v;
  }
  // ---- stage conv2 weights [tap][c] ----
  for (int idx = tid; idx < 288; idx += 256) {
    int tap = idx / 32, c = idx & 31;
    sW2[tap][c] = w2[c * 9 + tap];
  }
  // ---- stage feature tile (20x20, origin (oh-2, ow-2)); pad region -> 0 ----
  for (int idx = tid; idx < CIN_PAD * 480; idx += 256) {
    int c = idx / 480, r = idx % 480;
    int fy = r / 20, fx = r % 20;
    float v = 0.f;
    if (c < CIN && fy < 20) {
      int gy = oh - 2 + fy, gx = ow - 2 + fx;
      if (CIN == 1) {
        v = (gy < 0 || gy >= Hh || gx < 0 || gx >= Ww) ? 0.f : Ic[gy * Ww + gx];
      } else {
        v = feat_pair(c, gy, gx, Ic, In_, e0, e1, pr);
      }
    }
    sFeat[idx] = (_Float16)v;
  }
  __syncthreads();

  // ---- conv1 as implicit GEMM on the WMMA pipe ----
  const int wave = tid >> 5, lane = tid & 31;
  const int group = lane >> 4, half16 = lane & 15;

  // B fragments (32x16 f16): lane%16 = n, lane half selects K half.
  // n-major LDS layout -> each fragment = two aligned 16B vector loads.
  v16h bfrag[2][NCHUNK];
  #pragma unroll
  for (int nt = 0; nt < 2; ++nt)
    #pragma unroll
    for (int ck = 0; ck < NCHUNK; ++ck) {
      const v8h* wp = (const v8h*)&sW1T[16 * nt + half16][32 * ck + 16 * group];
      v8h lo = wp[0], hi = wp[1];
      bfrag[nt][ck] = __builtin_shufflevector(lo, hi,
          0, 1, 2, 3, 4, 5, 6, 7, 8, 9, 10, 11, 12, 13, 14, 15);
    }

  // hidden tile: 18*18 = 324 pixels, padded to 24 M-tiles of 16 (384 px):
  // each of the 8 waves owns exactly 3 M-tiles (wave, wave+8, wave+16).
  {
    const int p0 = (wave)      * 16 + half16;
    const int p1 = (wave + 8)  * 16 + half16;
    const int p2 = (wave + 16) * 16 + half16;
    const _Float16* f0 = sFeat + ((p0 / 18) * 20 + (p0 % 18));
    const _Float16* f1 = sFeat + ((p1 / 18) * 20 + (p1 % 18));
    const _Float16* f2 = sFeat + ((p2 / 18) * 20 + (p2 % 18));

    v8f a00 = {}, a10 = {}, a20 = {};   // N-tile 0 (channels 0-15)
    v8f a01 = {}, a11 = {}, a21 = {};   // N-tile 1 (channels 16-31)

    #pragma unroll
    for (int ck = 0; ck < NCHUNK; ++ck) {
      v16h x0, x1, x2;
      #pragma unroll
      for (int e = 0; e < 16; ++e) {
        // A k order chosen here (same as B): lanes 0-15 K in {0-15}, 16-31 {16-31}
        const int kb = 32 * ck + e;
        const int o  = group ? foff(kb + 16) : foff(kb);  // one cndmask, shared
        x0[e] = f0[o];
        x1[e] = f1[o];
        x2[e] = f2[o];
      }
      // 6 independent accumulator chains -> no hazard nops between WMMAs
      a00 = __builtin_amdgcn_wmma_f32_16x16x32_f16(false, x0, false, bfrag[0][ck],
                                                   (short)0, a00, false, false);
      a10 = __builtin_amdgcn_wmma_f32_16x16x32_f16(false, x1, false, bfrag[0][ck],
                                                   (short)0, a10, false, false);
      a20 = __builtin_amdgcn_wmma_f32_16x16x32_f16(false, x2, false, bfrag[0][ck],
                                                   (short)0, a20, false, false);
      a01 = __builtin_amdgcn_wmma_f32_16x16x32_f16(false, x0, false, bfrag[1][ck],
                                                   (short)0, a01, false, false);
      a11 = __builtin_amdgcn_wmma_f32_16x16x32_f16(false, x1, false, bfrag[1][ck],
                                                   (short)0, a11, false, false);
      a21 = __builtin_amdgcn_wmma_f32_16x16x32_f16(false, x2, false, bfrag[1][ck],
                                                   (short)0, a21, false, false);
    }

    // C layout: lane%16 = n, reg r -> m = 8*(lane/16)+r ; ReLU, store to LDS.
    // sHid padded to 384 pixels -> stores unconditional.
    _Float16* h0 = sHid + (size_t)((wave)      * 16 + 8 * group) * 32 + half16;
    _Float16* h1 = sHid + (size_t)((wave + 8)  * 16 + 8 * group) * 32 + half16;
    _Float16* h2 = sHid + (size_t)((wave + 16) * 16 + 8 * group) * 32 + half16;
    #pragma unroll
    for (int r = 0; r < 8; ++r) {
      h0[r * 32]      = (_Float16)relu_(a00[r]);
      h0[r * 32 + 16] = (_Float16)relu_(a01[r]);
      h1[r * 32]      = (_Float16)relu_(a10[r]);
      h1[r * 32 + 16] = (_Float16)relu_(a11[r]);
      h2[r * 32]      = (_Float16)relu_(a20[r]);
      h2[r * 32 + 16] = (_Float16)relu_(a21[r]);
    }
  }
  __syncthreads();

  // ---- conv2: 32->1 dot product from LDS, one pixel per thread ----
  {
    int ty = tid >> 4, tx = tid & 15;
    float s = 0.f;
    #pragma unroll
    for (int tap = 0; tap < 9; ++tap) {
      int pix = (ty + tap / 3) * 18 + (tx + tap % 3);
      const v8h* hv = (const v8h*)&sHid[pix * 32];   // 4x aligned b128 loads
      const float* wp = &sW2[tap][0];
      v8h h0 = hv[0], h1 = hv[1], h2 = hv[2], h3 = hv[3];
      #pragma unroll
      for (int j = 0; j < 8; ++j) {
        s += wp[j]      * (float)h0[j];
        s += wp[j + 8]  * (float)h1[j];
        s += wp[j + 16] * (float)h2[j];
        s += wp[j + 24] * (float)h3[j];
      }
    }
    out[(size_t)(oh + ty) * Ww + (ow + tx)] = s;
  }
}

// Gradient-descent update: alpha = 0.1*relu(conv1x1x3x3(fp)+b);
// dir = 2(fp - init) -/+ 2*D ; fp_next = fp - alpha*dir
__global__ __launch_bounds__(256) void update_fp(
    const float* __restrict__ fp, const float* __restrict__ finit,
    const float* __restrict__ res_t, const float* __restrict__ w_alpha,
    const float* __restrict__ b_alpha, float* __restrict__ fp_next)
{
  int idx = blockIdx.x * 256 + threadIdx.x;
  if (idx >= FP_ELEMS) return;
  int wq = idx & (Ww - 1);
  int hq = (idx >> 8) & (Hh - 1);
  int plane_idx = idx >> 16;            // b*N + n
  int n = plane_idx % Nn;
  int b = plane_idx / Nn;
  const float* pl = fp + (size_t)plane_idx * PLANE;

  float acc = 0.f;
  #pragma unroll
  for (int ty = 0; ty < 3; ++ty) {
    int gy = hq + ty - 1;
    if (gy < 0 || gy >= Hh) continue;
    #pragma unroll
    for (int tx = 0; tx < 3; ++tx) {
      int gx = wq + tx - 1;
      if (gx < 0 || gx >= Ww) continue;
      acc += w_alpha[ty * 3 + tx] * pl[gy * Ww + gx];
    }
  }
  acc += b_alpha[0];
  float alpha = 0.1f * (acc > 0.f ? acc : 0.f);

  float f = fp[idx];
  float dir = 2.0f * (f - finit[idx]);                  // LAMBDA_CONS = 1
  const float* rb = res_t + (size_t)b * RES_BSTRIDE + (size_t)hq * Ww + wq;
  if (n >= 1) {
    float Dm = fp[idx - PLANE] + rb[(size_t)(n - 1) * PLANE] - f;
    dir -= 2.0f * Dm;
  }
  if (n <= Nn - 2) {
    float Dp = f + rb[(size_t)n * PLANE] - fp[idx + PLANE];
    dir += 2.0f * Dp;
  }
  fp_next[idx] = f - alpha * dir;
}

extern "C" void kernel_launch(void* const* d_in, const int* in_sizes, int n_in,
                              void* d_out, int out_size, void* d_ws, size_t ws_size,
                              hipStream_t stream) {
  (void)in_sizes; (void)n_in; (void)out_size; (void)ws_size;
  const float* finit    = (const float*)d_in[0];
  const float* ev       = (const float*)d_in[1];
  const float* w_alpha  = (const float*)d_in[2];
  const float* b_alpha  = (const float*)d_in[3];
  const float* w1_first = (const float*)d_in[4];
  const float* w2_first = (const float*)d_in[5];
  const float* w1_other = (const float*)d_in[6];
  const float* w2_other = (const float*)d_in[7];
  const float* w1_final = (const float*)d_in[8];
  const float* w2_final = (const float*)d_in[9];

  float* out_final = (float*)d_out;
  float* resbase   = out_final + FP_ELEMS;   // residual (B,7,13,H,W)

  float* fpA = (float*)d_ws;
  float* fpB = fpA + FP_ELEMS;
  float* cur = fpA; float* nxt = fpB;

  hipMemcpyAsync(cur, finit, (size_t)FP_ELEMS * sizeof(float),
                 hipMemcpyDeviceToDevice, stream);

  dim3 block(256);
  dim3 cgrid(Ww / 16, Hh / 16, Bb);
  int ublocks = (FP_ELEMS + 255) / 256;

  for (int t = 0; t < Tt; ++t) {
    for (int i = 0; i < NP; ++i) {
      float* res_ti = resbase + ((size_t)t * NP + i) * PLANE;
      if (i == 0) {
        fused_conv<8><<<cgrid, block, 0, stream>>>(
            cur, ev, (const float*)nullptr, res_ti, w1_first, w2_first,
            i, RES_BSTRIDE);
      } else {
        const float* prev_ti = resbase + ((size_t)t * NP + (i - 1)) * PLANE;
        fused_conv<9><<<cgrid, block, 0, stream>>>(
            cur, ev, prev_ti, res_ti, w1_other, w2_other, i, RES_BSTRIDE);
      }
    }
    update_fp<<<dim3(ublocks), block, 0, stream>>>(
        cur, finit, resbase + (size_t)t * NP * PLANE, w_alpha, b_alpha, nxt);
    float* tmp = cur; cur = nxt; nxt = tmp;
  }

  dim3 fgrid(Ww / 16, Hh / 16, Bb * Nn);
  fused_conv<1><<<fgrid, block, 0, stream>>>(
      cur, (const float*)nullptr, (const float*)nullptr, out_final,
      w1_final, w2_final, 0, 0);
}
